// scMPNN_84954453115394
// MI455X (gfx1250) — matrix operand order repaired and checked by reading, not compiled, not run on hardware
//
#include <hip/hip_runtime.h>
#include <hip/hip_bf16.h>

typedef __attribute__((ext_vector_type(16))) __bf16 v16bf;
typedef __attribute__((ext_vector_type(8)))  float  v8f;

#define LN_EPS 1e-5f

__device__ __forceinline__ float gelu_f(float x){
  return 0.5f * x * (1.0f + erff(x * 0.7071067811865476f));
}

__device__ __forceinline__ v8f wmma_bf16(v16bf a, v16bf b, v8f c){
  // D = A(16x32 bf16) * B(32x16 bf16) + C(16x16 f32)
  return __builtin_amdgcn_wmma_f32_16x16x32_bf16(false, a, false, b, (short)0, c, false, false);
}

// A fragment from LDS tile (row-major, 'stride' bf16 elements per row).
// A 16x32, contiguous k order used here (B matches): lanes 0-15 -> M, elems 0..15 = K(kb..kb+15),
// kb = 0 (lanes 0-15) or 16 (lanes 16-31).
__device__ __forceinline__ v16bf load_a_lds(const __bf16* As, int stride, int lane, int k0){
  const int m  = lane & 15;
  const int kb = (lane >> 4) << 4;
  const __bf16* p = As + m * stride + k0 + kb;
  v16bf a;
#pragma unroll
  for(int e=0;e<16;++e){ a[e] = p[e]; }
  return a;
}

// B fragment from global pre-transposed weights WT[dout][din] (bf16).
// B 32x16, same contiguous k order as A: lane = N column, half-wave selects K 0..15 vs 16..31 -> 16 contiguous bf16.
__device__ __forceinline__ v16bf load_b_gl(const __bf16* WT, int Kdim, int ncol, int lane, int k0){
  const int n  = ncol + (lane & 15);
  const int kb = (lane >> 4) << 4;
  return *(const v16bf*)(WT + (long)n * Kdim + k0 + kb);
}

// ---------- weight prep: w[din][dout] f32 -> wt[dout_pad][din] bf16 (transpose+cast, zero pad)
__global__ void k_prep_w(const float* __restrict__ w, __bf16* __restrict__ wt,
                         int din, int dout, int dout_pad){
  int i = blockIdx.x * 256 + threadIdx.x;
  int total = dout_pad * din;
  if(i >= total) return;
  int n = i / din, k = i - n * din;
  float v = (n < dout) ? w[(long)k * dout + n] : 0.0f;
  wt[i] = (__bf16)v;
}

// ---------- generic row-tile GEMM: OUT[r][c] = act(A[r][:]@W + bias)  (16 rows / block)
__global__ __launch_bounds__(256) void k_gemm_rows(
    const float* __restrict__ A, const __bf16* __restrict__ WT,
    const float* __restrict__ bias, float* __restrict__ OUT,
    int Kdim, int Ndim, int Npad, int out_stride, int gelu_flag)
{
  __shared__ __bf16 As[16 * 392];
  const int S = 392;
  const int tid = threadIdx.x;
  const long row0 = (long)blockIdx.x * 16;
  for(int idx = tid; idx < 16 * Kdim; idx += 256){
    int r = idx / Kdim, c = idx - r * Kdim;
    As[r * S + c] = (__bf16)A[(row0 + r) * Kdim + c];
  }
  __syncthreads();
  const int wave = tid >> 5, lane = tid & 31;
  for(int nt = wave * 16; nt < Npad; nt += 128){
    v8f acc = {};
    for(int k0 = 0; k0 < Kdim; k0 += 32)
      acc = wmma_bf16(load_a_lds(As, S, lane, k0), load_b_gl(WT, Kdim, nt, lane, k0), acc);
    int col = nt + (lane & 15);
    if(col < Ndim){
      float bv = bias[col];
#pragma unroll
      for(int v = 0; v < 8; ++v){
        int m = v + ((lane >> 4) << 3);
        float x = acc[v] + bv;
        if(gelu_flag) x = gelu_f(x);
        OUT[(row0 + m) * out_stride + col] = x;
      }
    }
  }
}

// ---------- fused per-node message pass + node update (encoder first half & decoder)
// A = [hV_i | hE_ik | hV_{E_idx(i,k)}]  (16x384), msg = W3(gelu(W2(gelu(W1 A)))),
// t = LN1(hV + sum_k msg /16); hV_out = LN2(t + mix(t))
__global__ __launch_bounds__(256) void k_node_msg(
    const float* __restrict__ hV, const float* __restrict__ hE, const int* __restrict__ Eidx,
    const __bf16* __restrict__ WT1, const float* __restrict__ b1,
    const __bf16* __restrict__ WT2, const float* __restrict__ b2,
    const __bf16* __restrict__ WT3, const float* __restrict__ b3,
    const float* __restrict__ g1, const float* __restrict__ bb1,
    const float* __restrict__ g2, const float* __restrict__ bb2,
    const float* __restrict__ winw, const float* __restrict__ winb,
    const float* __restrict__ woutw, const float* __restrict__ woutb,
    float* __restrict__ hVout, int Nper)
{
  __shared__ __bf16 As[16 * 392];
  __shared__ __bf16 A1[16 * 136];
  __shared__ __bf16 A2[16 * 136];
  __shared__ float  csum[2][128];
  __shared__ float  tv[128], tv2[128], hid[256];
  __shared__ int    nbr[16];

  const int tid = threadIdx.x;
  const int g   = blockIdx.x;
  const int b   = g / Nper;
  const long base = (long)g * 128;

  if(tid < 16) nbr[tid] = b * Nper + Eidx[(long)g * 16 + tid];
  __syncthreads();

  for(int idx = tid; idx < 16 * 384; idx += 256){
    int r = idx / 384, c = idx - r * 384;
    float v;
    if(c < 128)      v = hV[base + c];
    else if(c < 256) v = hE[((long)g * 16 + r) * 128 + (c - 128)];
    else             v = hV[(long)nbr[r] * 128 + (c - 256)];
    As[r * 392 + c] = (__bf16)v;
  }
  __syncthreads();

  const int wave = tid >> 5, lane = tid & 31;
  const int col  = wave * 16 + (lane & 15);
  const int mofs = (lane >> 4) << 3;

  { // GEMM1: 384 -> 128, gelu
    v8f acc = {};
    for(int k0 = 0; k0 < 384; k0 += 32)
      acc = wmma_bf16(load_a_lds(As, 392, lane, k0), load_b_gl(WT1, 384, wave * 16, lane, k0), acc);
    float bv = b1[col];
#pragma unroll
    for(int v = 0; v < 8; ++v) A1[(v + mofs) * 136 + col] = (__bf16)gelu_f(acc[v] + bv);
  }
  __syncthreads();
  { // GEMM2: 128 -> 128, gelu
    v8f acc = {};
    for(int k0 = 0; k0 < 128; k0 += 32)
      acc = wmma_bf16(load_a_lds(A1, 136, lane, k0), load_b_gl(WT2, 128, wave * 16, lane, k0), acc);
    float bv = b2[col];
#pragma unroll
    for(int v = 0; v < 8; ++v) A2[(v + mofs) * 136 + col] = (__bf16)gelu_f(acc[v] + bv);
  }
  __syncthreads();
  { // GEMM3: 128 -> 128, + column sums over the 16 edges
    v8f acc = {};
    for(int k0 = 0; k0 < 128; k0 += 32)
      acc = wmma_bf16(load_a_lds(A2, 136, lane, k0), load_b_gl(WT3, 128, wave * 16, lane, k0), acc);
    float bv = b3[col], s = 0.f;
#pragma unroll
    for(int v = 0; v < 8; ++v) s += acc[v] + bv;
    csum[lane >> 4][col] = s;
  }
  __syncthreads();

  if(tid < 128){
    float ms = (csum[0][tid] + csum[1][tid]) * (1.0f / 16.0f);
    tv[tid] = hV[base + tid] + ms;
  }
  __syncthreads();
  if(tid < 128){ // LN1 (broadcast LDS reads are conflict-free)
    float m = 0.f; for(int i = 0; i < 128; ++i) m += tv[i]; m *= (1.0f / 128.0f);
    float va = 0.f; for(int i = 0; i < 128; ++i){ float d = tv[i] - m; va += d * d; } va *= (1.0f / 128.0f);
    tv2[tid] = (tv[tid] - m) * rsqrtf(va + LN_EPS) * g1[tid] + bb1[tid];
  }
  __syncthreads();
  { // mix hidden: 128 -> 256, gelu
    float s = winb[tid];
    for(int i = 0; i < 128; ++i) s += tv2[i] * winw[i * 256 + tid];
    hid[tid] = gelu_f(s);
  }
  __syncthreads();
  if(tid < 128){ // mix out: 256 -> 128, residual
    float s = woutb[tid];
    for(int i = 0; i < 256; ++i) s += hid[i] * woutw[i * 128 + tid];
    tv[tid] = tv2[tid] + s;
  }
  __syncthreads();
  if(tid < 128){ // LN2
    float m = 0.f; for(int i = 0; i < 128; ++i) m += tv[i]; m *= (1.0f / 128.0f);
    float va = 0.f; for(int i = 0; i < 128; ++i){ float d = tv[i] - m; va += d * d; } va *= (1.0f / 128.0f);
    hVout[base + tid] = (tv[tid] - m) * rsqrtf(va + LN_EPS) * g2[tid] + bb2[tid];
  }
}

// ---------- fused per-node edge update (encoder second half): hE = LN3(hE + msg)
__global__ __launch_bounds__(256) void k_edge_msg(
    const float* __restrict__ hV, float* __restrict__ hE, const int* __restrict__ Eidx,
    const __bf16* __restrict__ WT1, const float* __restrict__ b1,
    const __bf16* __restrict__ WT2, const float* __restrict__ b2,
    const __bf16* __restrict__ WT3, const float* __restrict__ b3,
    const float* __restrict__ g3, const float* __restrict__ bb3, int Nper)
{
  __shared__ __bf16 As[16 * 392];
  __shared__ __bf16 A1[16 * 136];
  __shared__ __bf16 A2[16 * 136];
  __shared__ float  Mbuf[16 * 128];
  __shared__ float  mrow[16], vrow[16];
  __shared__ int    nbr[16];

  const int tid = threadIdx.x;
  const int g   = blockIdx.x;
  const int b   = g / Nper;
  const long base = (long)g * 128;

  if(tid < 16) nbr[tid] = b * Nper + Eidx[(long)g * 16 + tid];
  __syncthreads();

  for(int idx = tid; idx < 16 * 384; idx += 256){
    int r = idx / 384, c = idx - r * 384;
    float v;
    if(c < 128)      v = hV[base + c];
    else if(c < 256) v = hE[((long)g * 16 + r) * 128 + (c - 128)];
    else             v = hV[(long)nbr[r] * 128 + (c - 256)];
    As[r * 392 + c] = (__bf16)v;
  }
  __syncthreads();

  const int wave = tid >> 5, lane = tid & 31;
  const int col  = wave * 16 + (lane & 15);
  const int mofs = (lane >> 4) << 3;

  {
    v8f acc = {};
    for(int k0 = 0; k0 < 384; k0 += 32)
      acc = wmma_bf16(load_a_lds(As, 392, lane, k0), load_b_gl(WT1, 384, wave * 16, lane, k0), acc);
    float bv = b1[col];
#pragma unroll
    for(int v = 0; v < 8; ++v) A1[(v + mofs) * 136 + col] = (__bf16)gelu_f(acc[v] + bv);
  }
  __syncthreads();
  {
    v8f acc = {};
    for(int k0 = 0; k0 < 128; k0 += 32)
      acc = wmma_bf16(load_a_lds(A1, 136, lane, k0), load_b_gl(WT2, 128, wave * 16, lane, k0), acc);
    float bv = b2[col];
#pragma unroll
    for(int v = 0; v < 8; ++v) A2[(v + mofs) * 136 + col] = (__bf16)gelu_f(acc[v] + bv);
  }
  __syncthreads();
  {
    v8f acc = {};
    for(int k0 = 0; k0 < 128; k0 += 32)
      acc = wmma_bf16(load_a_lds(A2, 136, lane, k0), load_b_gl(WT3, 128, wave * 16, lane, k0), acc);
    float bv = b3[col];
#pragma unroll
    for(int v = 0; v < 8; ++v) Mbuf[(v + mofs) * 128 + col] = acc[v] + bv;
  }
  __syncthreads();
  for(int idx = tid; idx < 16 * 128; idx += 256){
    int r = idx >> 7, c = idx & 127;
    Mbuf[idx] += hE[((long)g * 16 + r) * 128 + c];
  }
  __syncthreads();
  if(tid < 16){
    float m = 0.f; for(int i = 0; i < 128; ++i) m += Mbuf[tid * 128 + i]; m *= (1.0f / 128.0f);
    float va = 0.f; for(int i = 0; i < 128; ++i){ float d = Mbuf[tid * 128 + i] - m; va += d * d; }
    mrow[tid] = m; vrow[tid] = rsqrtf(va * (1.0f / 128.0f) + LN_EPS);
  }
  __syncthreads();
  for(int idx = tid; idx < 16 * 128; idx += 256){
    int r = idx >> 7, c = idx & 127;
    hE[((long)g * 16 + r) * 128 + c] = (Mbuf[idx] - mrow[r]) * vrow[r] * g3[c] + bb3[c];
  }
}

// ---------- cell head: per-node  cellh = gelu( sum_k gelu(hEV_k . wc1 + bc1) * wc2_k + bc2 )
__global__ __launch_bounds__(256) void k_cell12(
    const float* __restrict__ hV, const float* __restrict__ hE, const int* __restrict__ Eidx,
    const float* __restrict__ w1, const float* __restrict__ b1,
    const float* __restrict__ w2, const float* __restrict__ b2,
    float* __restrict__ cellh, int Nper)
{
  __shared__ float red[256];
  __shared__ int nbr[16];
  const int tid = threadIdx.x;
  const int g = blockIdx.x;
  const int b = g / Nper;
  const long base = (long)g * 128;
  if(tid < 16) nbr[tid] = b * Nper + Eidx[(long)g * 16 + tid];
  __syncthreads();
  const int k = tid >> 4, sub = tid & 15;
  float s = 0.f;
  for(int c = sub; c < 384; c += 16){
    float v;
    if(c < 128)      v = hV[base + c];
    else if(c < 256) v = hE[((long)g * 16 + k) * 128 + (c - 128)];
    else             v = hV[(long)nbr[k] * 128 + (c - 256)];
    s += v * w1[c];
  }
  red[tid] = s; __syncthreads();
  if(sub < 8) red[tid] += red[tid + 8]; __syncthreads();
  if(sub < 4) red[tid] += red[tid + 4]; __syncthreads();
  if(sub < 2) red[tid] += red[tid + 2]; __syncthreads();
  if(sub == 0) red[tid] = gelu_f(red[tid] + red[tid + 1] + b1[0]) * w2[k];
  __syncthreads();
  if(tid == 0){
    float t = b2[0];
    for(int kk = 0; kk < 16; ++kk) t += red[kk * 16];
    cellh[g] = gelu_f(t);
  }
}

__global__ __launch_bounds__(256) void k_cell3(
    const float* __restrict__ cellh, const float* __restrict__ w3, const float* __restrict__ b3,
    float* __restrict__ out, int Nper)
{
  __shared__ float red[256];
  const int b = blockIdx.x / 9, j = blockIdx.x % 9;
  float s = 0.f;
  for(int n = threadIdx.x; n < Nper; n += 256) s += cellh[b * Nper + n] * w3[n * 9 + j];
  red[threadIdx.x] = s; __syncthreads();
  for(int off = 128; off > 0; off >>= 1){
    if(threadIdx.x < off) red[threadIdx.x] += red[threadIdx.x + off];
    __syncthreads();
  }
  if(threadIdx.x == 0) out[b * 9 + j] = red[0] + b3[j];
}

// =======================================================================================
extern "C" void kernel_launch(void* const* d_in, const int* in_sizes, int n_in,
                              void* d_out, int out_size, void* d_ws, size_t ws_size,
                              hipStream_t stream) {
  (void)in_sizes; (void)n_in; (void)out_size; (void)ws_size;
  const int BN  = 2 * 8000;       // node rows
  const int ER  = BN * 16;        // edge rows
  const float* V    = (const float*)d_in[0];
  const float* E    = (const float*)d_in[1];
  const int*   Eidx = (const int*)d_in[2];    // JAX x64 off -> int32
  auto F = [&](int i){ return (const float*)d_in[i]; };

  char* ws = (char*)d_ws; size_t off = 0;
  auto carve = [&](size_t bytes){ void* p = ws + off; off = (off + bytes + 255) & ~(size_t)255; return p; };
  float* hVa   = (float*)carve((size_t)BN * 128 * 4);
  float* hVb   = (float*)carve((size_t)BN * 128 * 4);
  float* hE    = (float*)carve((size_t)ER * 128 * 4);
  float* cellh = (float*)carve((size_t)BN * 4);
  auto carve_w = [&](int din, int dpad){ return (__bf16*)carve((size_t)din * dpad * 2); };

  __bf16* wtv  = carve_w(128, 128);
  __bf16* wte  = carve_w( 64, 128);
  __bf16* eW1[2], *eW2[2], *eW3[2], *eW11[2], *eW12[2], *eW13[2];
  for(int l = 0; l < 2; ++l){
    eW1[l]  = carve_w(384, 128); eW2[l]  = carve_w(128, 128); eW3[l]  = carve_w(128, 128);
    eW11[l] = carve_w(384, 128); eW12[l] = carve_w(128, 128); eW13[l] = carve_w(128, 128);
  }
  __bf16* dW1 = carve_w(384, 128); __bf16* dW2 = carve_w(128, 128); __bf16* dW3 = carve_w(128, 128);
  __bf16* wtrec = carve_w(128, 208);

  auto prep = [&](int widx, __bf16* wt, int din, int dout, int dpad){
    int total = dpad * din;
    k_prep_w<<<dim3((total + 255) / 256), dim3(256), 0, stream>>>(F(widx), wt, din, dout, dpad);
  };
  // param leaf indices (setup_inputs insertion order)
  prep(3,  wtv, 128, 128, 128);
  prep(5,  wte,  64, 128, 128);
  for(int l = 0; l < 2; ++l){
    int bi = 7 + l * 22;
    prep(bi + 0,  eW1[l],  384, 128, 128);
    prep(bi + 2,  eW2[l],  128, 128, 128);
    prep(bi + 4,  eW3[l],  128, 128, 128);
    prep(bi + 6,  eW11[l], 384, 128, 128);
    prep(bi + 8,  eW12[l], 128, 128, 128);
    prep(bi + 10, eW13[l], 128, 128, 128);
  }
  prep(51, dW1, 384, 128, 128);
  prep(53, dW2, 128, 128, 128);
  prep(55, dW3, 128, 128, 128);
  prep(65, wtrec, 128, 200, 208);

  // embeddings
  k_gemm_rows<<<dim3(BN / 16), dim3(256), 0, stream>>>(V, wtv, F(4), hVa, 128, 128, 128, 128, 0);
  k_gemm_rows<<<dim3(ER / 16), dim3(256), 0, stream>>>(E, wte, F(6), hE, 64, 128, 128, 128, 0);

  // encoder layers
  float* cur = hVa; float* nxt = hVb;
  for(int l = 0; l < 2; ++l){
    int bi = 7 + l * 22;
    k_node_msg<<<dim3(BN), dim3(256), 0, stream>>>(
        cur, hE, Eidx,
        eW1[l], F(bi + 1), eW2[l], F(bi + 3), eW3[l], F(bi + 5),
        F(bi + 12), F(bi + 13), F(bi + 14), F(bi + 15),
        F(bi + 18), F(bi + 19), F(bi + 20), F(bi + 21),
        nxt, 8000);
    float* t = cur; cur = nxt; nxt = t;
    k_edge_msg<<<dim3(BN), dim3(256), 0, stream>>>(
        cur, hE, Eidx,
        eW11[l], F(bi + 7), eW12[l], F(bi + 9), eW13[l], F(bi + 11),
        F(bi + 16), F(bi + 17), 8000);
  }

  // cell-state head
  k_cell12<<<dim3(BN), dim3(256), 0, stream>>>(cur, hE, Eidx, F(67), F(68), F(69), F(70), cellh, 8000);
  k_cell3<<<dim3(18), dim3(256), 0, stream>>>(cellh, F(71), F(72), (float*)d_out + 3200000, 8000);

  // decoder layer
  k_node_msg<<<dim3(BN), dim3(256), 0, stream>>>(
      cur, hE, Eidx,
      dW1, F(52), dW2, F(54), dW3, F(56),
      F(57), F(58), F(59), F(60),
      F(61), F(62), F(63), F(64),
      nxt, 8000);
  { float* t = cur; cur = nxt; nxt = t; }

  // recover head: 128 -> 200 (padded to 208)
  k_gemm_rows<<<dim3(BN / 16), dim3(256), 0, stream>>>(cur, wtrec, F(66), (float*)d_out, 128, 200, 208, 200, 0);
}
